// TextEncoder_36498632081761
// MI455X (gfx1250) — matrix operand order repaired and checked by reading, not compiled, not run on hardware
//
#include <hip/hip_runtime.h>
#include <hip/hip_bf16.h>

// ---------------------------------------------------------------------------
// CDNA5 (gfx1250) text-encoder forward.
// GEMMs: LDS-staged 64x64 macro-tiles, 4 waves/block, each wave a 32x32 C
// block = 4x V_WMMA_F32_16X16X32_BF16 per 32-wide k-step (fp32 accumulate).
// Double-buffered GLOBAL_LOAD_ASYNC_TO_LDS_B32 staging overlaps global
// traffic with the matrix pipes.  Workspace requirement: ~143 MB.
// ---------------------------------------------------------------------------

#define BB   8
#define TT   1024
#define HH   192
#define NHH  2
#define LL   6
#define FFNN 768
#define OUTC 192
#define WINW 4
#define KCC  96          // HH / NHH
#define LDW  33          // LDS row stride in floats (odd -> conflict-free)

#if defined(__has_builtin)
#if __has_builtin(__builtin_amdgcn_global_load_async_to_lds_b32)
#define TRY_ASYNC 1
#endif
#endif

typedef __attribute__((ext_vector_type(16))) __bf16 v16bf;
typedef __attribute__((ext_vector_type(8)))  float  v8f;

// fp32 -> bf16 via native v_cvt (RTNE)
__device__ __forceinline__ __bf16 f2bf(float f) { return (__bf16)f; }

// one fp32 element: global -> LDS (async if available)
__device__ __forceinline__ void stage1(const float* src, float* dstLds) {
#if TRY_ASYNC
    __builtin_amdgcn_global_load_async_to_lds_b32(
        (__attribute__((address_space(1))) int*)(void*)const_cast<float*>(src),
        (__attribute__((address_space(3))) int*)(void*)dstLds, 0, 0);
#else
    *dstLds = *src;
#endif
}
__device__ __forceinline__ void stage_fence() {
#if TRY_ASYNC
    asm volatile("s_wait_asynccnt 0x0" ::: "memory");
#endif
    __syncthreads();
}

// stage 64 rows x 32 k of a row-major matrix (ld = row stride) into LDS
__device__ __forceinline__ void stage_w(const float* W, int ld, int m0, int k0,
                                        int tid, float* dst) {
    for (int i = tid; i < 64 * 32; i += 128) {
        int kk = i & 31, mm = i >> 5;
        stage1(&W[(size_t)(m0 + mm) * ld + k0 + kk], &dst[mm * LDW + kk]);
    }
}
// stage 32 k x 64 n of a k-major activation (row stride TT), transposed to n-major
__device__ __forceinline__ void stage_xT(const float* Xb, int n0, int k0,
                                         int tid, float* dst) {
    for (int i = tid; i < 64 * 32; i += 128) {
        int nn = i & 63, kk = i >> 6;
        stage1(&Xb[(size_t)(k0 + kk) * TT + n0 + nn], &dst[nn * LDW + kk]);
    }
}

// A fragment: 16x32 subtile, row = LDS row of 32 k-floats (16-bit A layout)
__device__ __forceinline__ v16bf fragA(const float* row, int g) {
    v16bf a;
#pragma unroll
    for (int e = 0; e < 8; ++e) {
        a[e]     = f2bf(row[8 * g + e]);
        a[e + 8] = f2bf(row[16 + 8 * g + e]);
    }
    return a;
}
// B fragment: 32x16 subtile, row = LDS row (n-major) of 32 k-floats
__device__ __forceinline__ v16bf fragB(const float* row, int g) {
    v16bf b;
#pragma unroll
    for (int e = 0; e < 8; ++e) {
        b[e]     = f2bf(row[8 * g + e]);
        b[e + 8] = f2bf(row[16 + 8 * g + e]);
    }
    return b;
}

#define WMMA_BF16(c, a, b) \
    (c) = __builtin_amdgcn_wmma_f32_16x16x32_bf16(false, (a), false, (b), (short)0, (c), false, false)

// 4-WMMA register-blocked step: wave (wm,wn) computes 32x32 of the 64x64 tile
__device__ __forceinline__ void mma_step(const float* As, const float* Bs,
                                         int wm, int wn, int l, int g, v8f acc[2][2]) {
    v16bf a0 = fragA(As + (wm * 32 + l) * LDW, g);
    v16bf a1 = fragA(As + (wm * 32 + 16 + l) * LDW, g);
    v16bf b0 = fragB(Bs + (wn * 32 + l) * LDW, g);
    v16bf b1 = fragB(Bs + (wn * 32 + 16 + l) * LDW, g);
    WMMA_BF16(acc[0][0], a0, b0);
    WMMA_BF16(acc[0][1], a0, b1);
    WMMA_BF16(acc[1][0], a1, b0);
    WMMA_BF16(acc[1][1], a1, b1);
}

// ---------------------------------------------------------------------------
// Embedding + length mask
// ---------------------------------------------------------------------------
__global__ void k_embed(const int* __restrict__ tok, const int* __restrict__ lens,
                        const float* __restrict__ emb, float* __restrict__ X,
                        float* __restrict__ msk) {
    size_t i = (size_t)blockIdx.x * 256 + threadIdx.x;
    const size_t N = (size_t)BB * HH * TT;
    if (i >= N) return;
    int t = (int)(i % TT);
    size_t r = i / TT;
    int c = (int)(r % HH);
    int b = (int)(r / HH);
    float m = (t < lens[b]) ? 1.0f : 0.0f;
    int tk = tok[b * TT + t];
    X[i] = emb[(size_t)tk * HH + c] * 13.856406460551018f * m;   // sqrt(192)
    if (c == 0) msk[b * TT + t] = m;
}

// ---------------------------------------------------------------------------
// Dense staged GEMM: Y = W(MxC) @ X(b,C,T) + bias.  64x64 macro-tile,
// double-buffered async staging.  mode 0: Y[b,o,t]; mode 1: Y[b,h,t,kc]
// ---------------------------------------------------------------------------
__global__ __launch_bounds__(128) void k_gemm64(
    const float* __restrict__ W, const float* __restrict__ bias,
    const float* __restrict__ X, float* __restrict__ Y,
    int Kdim, int Mtot, float scale, int mode)
{
    __shared__ float As[2][64 * LDW];
    __shared__ float Bs[2][64 * LDW];
    const int tid = threadIdx.x;
    const int lane = tid & 31, l = lane & 15, g = lane >> 4;
    const int wave = tid >> 5, wm = wave >> 1, wn = wave & 1;
    const int n0 = blockIdx.x * 64;     // t
    const int m0 = blockIdx.y * 64;     // o
    const int b  = blockIdx.z;
    const float* Xb = X + (size_t)b * Kdim * TT;
    v8f acc[2][2] = {};
    const int nsteps = Kdim / 32;
    stage_w(W, Kdim, m0, 0, tid, As[0]);
    stage_xT(Xb, n0, 0, tid, Bs[0]);
    stage_fence();
    for (int s = 0; s < nsteps; ++s) {
        const int cur = s & 1;
        if (s + 1 < nsteps) {                       // prefetch next k-step
            stage_w(W, Kdim, m0, (s + 1) * 32, tid, As[cur ^ 1]);
            stage_xT(Xb, n0, (s + 1) * 32, tid, Bs[cur ^ 1]);
        }
        mma_step(As[cur], Bs[cur], wm, wn, l, g, acc);
        stage_fence();
    }
#pragma unroll
    for (int mi = 0; mi < 2; ++mi)
#pragma unroll
        for (int ni = 0; ni < 2; ++ni) {
            const int t = n0 + wn * 32 + ni * 16 + l;
#pragma unroll
            for (int r = 0; r < 8; ++r) {
                int o = m0 + wm * 32 + mi * 16 + r + 8 * g;
                float val = (acc[mi][ni][r] + bias[o]) * scale;
                if (mode == 0) {
                    Y[((size_t)b * Mtot + o) * TT + t] = val;
                } else {
                    int h = o / KCC, kc = o % KCC;
                    Y[(((size_t)b * NHH + h) * TT + t) * KCC + kc] = val;
                }
            }
        }
}

// ---------------------------------------------------------------------------
// K=3 "SAME" conv as staged GEMM (3 shifted K passes, masked input loader).
// ---------------------------------------------------------------------------
__global__ __launch_bounds__(128) void k_conv64(
    const float* __restrict__ W, const float* __restrict__ bias,
    const float* __restrict__ X, const float* __restrict__ msk,
    float* __restrict__ Y, int Cdim, int Mtot, int relu, int maskOut)
{
    __shared__ float As[64 * LDW];
    __shared__ float Bs[64 * LDW];
    const int tid = threadIdx.x;
    const int lane = tid & 31, l = lane & 15, g = lane >> 4;
    const int wave = tid >> 5, wm = wave >> 1, wn = wave & 1;
    const int n0 = blockIdx.x * 64;
    const int m0 = blockIdx.y * 64;
    const int b  = blockIdx.z;
    const float* Xb = X + (size_t)b * Cdim * TT;
    const float* mb = msk + (size_t)b * TT;
    v8f acc[2][2] = {};
    for (int j = 0; j < 3; ++j) {
        for (int k0 = 0; k0 < Cdim; k0 += 32) {
            for (int i = tid; i < 64 * 32; i += 128) {      // A: stride-3 weights (sync)
                int kk = i & 31, mm = i >> 5;
                As[mm * LDW + kk] = W[((size_t)(m0 + mm) * Cdim + k0 + kk) * 3 + j];
            }
            for (int i = tid; i < 64 * 32; i += 128) {      // B: shifted + masked input
                int nn = i & 63, kk = i >> 6;
                int tj = n0 + nn + j - 1;
                float v = 0.0f;
                if (tj >= 0 && tj < TT) v = Xb[(size_t)(k0 + kk) * TT + tj] * mb[tj];
                Bs[nn * LDW + kk] = v;
            }
            __syncthreads();
            mma_step(As, Bs, wm, wn, l, g, acc);
            __syncthreads();
        }
    }
#pragma unroll
    for (int mi = 0; mi < 2; ++mi)
#pragma unroll
        for (int ni = 0; ni < 2; ++ni) {
            const int t = n0 + wn * 32 + ni * 16 + l;
            const float mo = maskOut ? mb[t] : 1.0f;
#pragma unroll
            for (int r = 0; r < 8; ++r) {
                int o = m0 + wm * 32 + mi * 16 + r + 8 * g;
                float v = acc[mi][ni][r] + bias[o];
                if (relu) v = fmaxf(v, 0.0f);
                Y[((size_t)b * Mtot + o) * TT + t] = v * mo;
            }
        }
}

// ---------------------------------------------------------------------------
// scores[z,t,s] = qs.k + rel_k band, masked.  Staged 64x64 (K = KC = 96),
// double-buffered async staging.
// ---------------------------------------------------------------------------
__global__ __launch_bounds__(128) void k_scores64(
    const float* __restrict__ Q, const float* __restrict__ Km,
    const float* __restrict__ relk, const int* __restrict__ lens,
    float* __restrict__ S)
{
    __shared__ float As[2][64 * LDW];
    __shared__ float Bs[2][64 * LDW];
    const int tid = threadIdx.x;
    const int lane = tid & 31, l = lane & 15, g = lane >> 4;
    const int wave = tid >> 5, wm = wave >> 1, wn = wave & 1;
    const int s0 = blockIdx.x * 64;
    const int t0 = blockIdx.y * 64;
    const int z  = blockIdx.z;
    const int b  = z / NHH;
    const float* Qz = Q  + (size_t)z * TT * KCC;
    const float* Kz = Km + (size_t)z * TT * KCC;
    v8f acc[2][2] = {};
    const int nsteps = KCC / 32;
    stage_w(Qz, KCC, t0, 0, tid, As[0]);   // A = Q rows (row-major)
    stage_w(Kz, KCC, s0, 0, tid, Bs[0]);   // B = K rows (row-major = n-major)
    stage_fence();
    for (int s = 0; s < nsteps; ++s) {
        const int cur = s & 1;
        if (s + 1 < nsteps) {
            stage_w(Qz, KCC, t0, (s + 1) * 32, tid, As[cur ^ 1]);
            stage_w(Kz, KCC, s0, (s + 1) * 32, tid, Bs[cur ^ 1]);
        }
        mma_step(As[cur], Bs[cur], wm, wn, l, g, acc);
        stage_fence();
    }
    const int len = lens[b];
#pragma unroll
    for (int mi = 0; mi < 2; ++mi)
#pragma unroll
        for (int ni = 0; ni < 2; ++ni) {
            const int s = s0 + wn * 32 + ni * 16 + l;
#pragma unroll
            for (int r = 0; r < 8; ++r) {
                int t = t0 + wm * 32 + mi * 16 + r + 8 * g;
                float v = acc[mi][ni][r];
                int d = s - t;
                if (d >= -WINW && d <= WINW) {
                    const float* qrow = Qz + (size_t)t * KCC;
                    const float* er   = relk + (size_t)(d + WINW) * KCC;
                    float add = 0.f;
                    for (int kk = 0; kk < KCC; ++kk) add += qrow[kk] * er[kk];
                    v += add;
                }
                if (t >= len || s >= len) v = -10000.0f;
                S[((size_t)z * TT + t) * TT + s] = v;
            }
        }
}

// ---------------------------------------------------------------------------
// In-place row softmax over T elements, one block per row.
// ---------------------------------------------------------------------------
__global__ __launch_bounds__(256) void k_softmax(float* __restrict__ S) {
    __shared__ float red[256];
    float* p = S + (size_t)blockIdx.x * TT;
    const int tid = threadIdx.x;
    float mx = -3.4e38f;
    for (int i = tid; i < TT; i += 256) mx = fmaxf(mx, p[i]);
    red[tid] = mx; __syncthreads();
    for (int st = 128; st > 0; st >>= 1) {
        if (tid < st) red[tid] = fmaxf(red[tid], red[tid + st]);
        __syncthreads();
    }
    mx = red[0]; __syncthreads();
    float sum = 0.f;
    for (int i = tid; i < TT; i += 256) { float e = __expf(p[i] - mx); p[i] = e; sum += e; }
    red[tid] = sum; __syncthreads();
    for (int st = 128; st > 0; st >>= 1) {
        if (tid < st) red[tid] += red[tid + st];
        __syncthreads();
    }
    const float inv = 1.0f / red[0];
    for (int i = tid; i < TT; i += 256) p[i] *= inv;
}

// ---------------------------------------------------------------------------
// O[b, h*KC+kc, t] = sum_s P[z,t,s] V[z,s,kc]  + rel_v band  (wave/tile, N=96)
// ---------------------------------------------------------------------------
__global__ __launch_bounds__(32) void k_attnout(
    const float* __restrict__ P, const float* __restrict__ V,
    const float* __restrict__ relv, float* __restrict__ O)
{
    const int lane = threadIdx.x, l = lane & 15, g = lane >> 4;
    const int n0 = blockIdx.x * 16;     // kc
    const int t0 = blockIdx.y * 16;     // t
    const int z  = blockIdx.z;
    const int b  = z / NHH, h = z % NHH;
    const float* Pz = P + (size_t)z * TT * TT;
    const float* Vz = V + (size_t)z * TT * KCC;
    v8f acc = {};
    for (int k0 = 0; k0 < TT; k0 += 32) {
        v16bf a, bf;
#pragma unroll
        for (int e = 0; e < 16; ++e) {
            int ka = (e & 7) + 8 * g + ((e >> 3) << 4);
            a[e]  = f2bf(Pz[(size_t)(t0 + l) * TT + k0 + ka]);
            bf[e] = f2bf(Vz[(size_t)(k0 + ka) * KCC + (n0 + l)]);
        }
        WMMA_BF16(acc, a, bf);
    }
#pragma unroll
    for (int r = 0; r < 8; ++r) {
        int t  = t0 + r + 8 * g;
        int kc = n0 + l;
        float v = acc[r];
        const float* prow = Pz + (size_t)t * TT;
#pragma unroll
        for (int d = -WINW; d <= WINW; ++d) {
            int s = t + d;
            if (s >= 0 && s < TT) v += prow[s] * relv[(size_t)(d + WINW) * KCC + kc];
        }
        O[((size_t)b * HH + h * KCC + kc) * TT + t] = v;
    }
}

// ---------------------------------------------------------------------------
// LayerNorm over channels: Xout = LN(Xin + Yin) * gamma + beta
// ---------------------------------------------------------------------------
__global__ __launch_bounds__(192) void k_ln(
    const float* __restrict__ Xin, const float* __restrict__ Yin,
    const float* __restrict__ gam, const float* __restrict__ bet,
    float* __restrict__ Xout)
{
    __shared__ float s1[192], s2[192], mv[2];
    const int idx = blockIdx.x;           // b*T + t
    const int b = idx / TT, t = idx % TT;
    const int c = threadIdx.x;
    const size_t off = ((size_t)b * HH + c) * TT + t;
    const float v = Xin[off] + Yin[off];
    s1[c] = v; s2[c] = v * v; __syncthreads();
    for (int st = 96; st >= 3; st >>= 1) {
        if (c < st) { s1[c] += s1[c + st]; s2[c] += s2[c + st]; }
        __syncthreads();
    }
    if (c == 0) {
        float sum = s1[0] + s1[1] + s1[2];
        float ssq = s2[0] + s2[1] + s2[2];
        float m = sum * (1.0f / HH);
        mv[0] = m;
        mv[1] = rsqrtf(ssq * (1.0f / HH) - m * m + 1e-5f);
    }
    __syncthreads();
    Xout[off] = (v - mv[0]) * mv[1] * gam[c] + bet[c];
}

__global__ void k_maskx(const float* __restrict__ X, const float* __restrict__ msk,
                        float* __restrict__ Y) {
    size_t i = (size_t)blockIdx.x * 256 + threadIdx.x;
    const size_t N = (size_t)BB * HH * TT;
    if (i >= N) return;
    int t = (int)(i % TT);
    int b = (int)(i / ((size_t)HH * TT));
    Y[i] = X[i] * msk[b * TT + t];
}

__global__ void k_outputs(const float* __restrict__ stats, const float* __restrict__ eps,
                          const float* __restrict__ msk, float* __restrict__ out) {
    const size_t Z = (size_t)BB * OUTC * TT;
    size_t i = (size_t)blockIdx.x * 256 + threadIdx.x;
    if (i >= Z) return;
    int t = (int)(i % TT);
    size_t r = i / TT;
    int o = (int)(r % OUTC);
    int b = (int)(r / OUTC);
    float mk = msk[b * TT + t];
    float m  = stats[((size_t)b * (2 * OUTC) + o) * TT + t] * mk;
    float lg = stats[((size_t)b * (2 * OUTC) + OUTC + o) * TT + t] * mk;
    out[i]         = m + eps[i] * __expf(lg) * mk;
    out[Z + i]     = m;
    out[2 * Z + i] = lg;
}

__global__ void k_writemask(const float* __restrict__ msk, float* __restrict__ out) {
    int i = blockIdx.x * 256 + threadIdx.x;
    if (i < BB * TT) out[i] = msk[i];
}

// ---------------------------------------------------------------------------
extern "C" void kernel_launch(void* const* d_in, const int* in_sizes, int n_in,
                              void* d_out, int out_size, void* d_ws, size_t ws_size,
                              hipStream_t stream) {
    const int*   tok   = (const int*)  d_in[0];
    const int*   lens  = (const int*)  d_in[1];
    const float* emb   = (const float*)d_in[2];
    const float* qw    = (const float*)d_in[3];
    const float* qb    = (const float*)d_in[4];
    const float* kw    = (const float*)d_in[5];
    const float* kb    = (const float*)d_in[6];
    const float* vw    = (const float*)d_in[7];
    const float* vb    = (const float*)d_in[8];
    const float* ow    = (const float*)d_in[9];
    const float* ob    = (const float*)d_in[10];
    const float* rel_k = (const float*)d_in[11];
    const float* rel_v = (const float*)d_in[12];
    const float* ln1g  = (const float*)d_in[13];
    const float* ln1b  = (const float*)d_in[14];
    const float* f1w   = (const float*)d_in[15];
    const float* f1b   = (const float*)d_in[16];
    const float* f2w   = (const float*)d_in[17];
    const float* f2b   = (const float*)d_in[18];
    const float* ln2g  = (const float*)d_in[19];
    const float* ln2b  = (const float*)d_in[20];
    const float* pw    = (const float*)d_in[21];
    const float* pb    = (const float*)d_in[22];
    const float* eps   = (const float*)d_in[23];

    float* out = (float*)d_out;
    const size_t BHT = (size_t)BB * HH * TT;           // 1572864
    const size_t Z   = (size_t)BB * OUTC * TT;         // == BHT

    // workspace layout (floats); ~35.66M floats (~143 MB)
    float* w     = (float*)d_ws;
    float* x     = w;                                   // (B,H,T)
    float* msk   = x + BHT;                             // (B,T)
    float* q     = msk + (size_t)BB * TT;               // (B,NH,T,KC)
    float* kbuf  = q + BHT;
    float* v     = kbuf + BHT;
    float* sc    = v + BHT;                             // (B*NH, T, T)
    float* y1    = sc + (size_t)BB * NHH * TT * TT;     // (B,FFN,T)
    float* obuf  = y1 + (size_t)BB * FFNN * TT;         // (B,H,T)
    float* ybuf  = obuf + BHT;                          // (B,H,T)
    float* stats = ybuf + BHT;                          // (B,2*OUT,T)

    const int eltBlocks = (int)((BHT + 255) / 256);
    const float qscale = 0.10206207261596575f;          // 1/sqrt(96)

    k_embed<<<eltBlocks, 256, 0, stream>>>(tok, lens, emb, x, msk);

    for (int i = 0; i < LL; ++i) {
        const float* qwi = qw + (size_t)i * HH * HH;
        const float* kwi = kw + (size_t)i * HH * HH;
        const float* vwi = vw + (size_t)i * HH * HH;
        const float* owi = ow + (size_t)i * HH * HH;
        const float* rki = rel_k + (size_t)i * (2 * WINW + 1) * KCC;
        const float* rvi = rel_v + (size_t)i * (2 * WINW + 1) * KCC;

        dim3 gp(TT / 64, HH / 64, BB);                  // (16,3,8)
        k_gemm64<<<gp, 128, 0, stream>>>(qwi, qb + i * HH, x, q,    HH, HH, qscale, 1);
        k_gemm64<<<gp, 128, 0, stream>>>(kwi, kb + i * HH, x, kbuf, HH, HH, 1.0f,   1);
        k_gemm64<<<gp, 128, 0, stream>>>(vwi, vb + i * HH, x, v,    HH, HH, 1.0f,   1);

        dim3 gs(TT / 64, TT / 64, BB * NHH);            // (16,16,16)
        k_scores64<<<gs, 128, 0, stream>>>(q, kbuf, rki, lens, sc);

        k_softmax<<<BB * NHH * TT, 256, 0, stream>>>(sc);

        dim3 ga(KCC / 16, TT / 16, BB * NHH);           // (6,64,16)
        k_attnout<<<ga, 32, 0, stream>>>(sc, v, rvi, obuf);

        k_gemm64<<<gp, 128, 0, stream>>>(owi, ob + i * HH, obuf, ybuf, HH, HH, 1.0f, 0);
        k_ln<<<BB * TT, 192, 0, stream>>>(x, ybuf, ln1g + i * HH, ln1b + i * HH, x);

        dim3 gf1(TT / 64, FFNN / 64, BB);               // (16,12,8)
        k_conv64<<<gf1, 128, 0, stream>>>(f1w + (size_t)i * FFNN * HH * 3, f1b + i * FFNN,
                                          x, msk, y1, HH, FFNN, 1, 0);
        dim3 gf2(TT / 64, HH / 64, BB);                 // (16,3,8)
        k_conv64<<<gf2, 128, 0, stream>>>(f2w + (size_t)i * HH * FFNN * 3, f2b + i * HH,
                                          y1, msk, ybuf, FFNN, HH, 0, 1);
        k_ln<<<BB * TT, 192, 0, stream>>>(x, ybuf, ln2g + i * HH, ln2b + i * HH, x);
    }

    k_maskx<<<eltBlocks, 256, 0, stream>>>(x, msk, out + 3 * Z);

    dim3 gpw(TT / 64, (2 * OUTC) / 64, BB);             // (16,6,8)
    k_gemm64<<<gpw, 128, 0, stream>>>(pw, pb, out + 3 * Z, stats, HH, 2 * OUTC, 1.0f, 0);

    k_outputs<<<(int)((Z + 255) / 256), 256, 0, stream>>>(stats, eps, msk, out);
    k_writemask<<<(BB * TT + 255) / 256, 256, 0, stream>>>(msk, out + 4 * Z);
}
